// AdaptiveFNO2d_26834955665754
// MI455X (gfx1250) — matrix operand order repaired and checked by reading, not compiled, not run on hardware
//
#include <hip/hip_runtime.h>
#include <math.h>

// ---------------------------------------------------------------------------
// FNO2d on gfx1250 (MI455X): factored truncated-DFT spectral conv + WMMA GEMMs.
// All heavy GEMMs use v_wmma_f32_16x16x32_bf16 with hi/lo split-bf16 operands
// (3 WMMAs per tile-step) for ~fp32 accuracy while staying on the matrix pipe.
// ---------------------------------------------------------------------------

typedef __bf16 bf16_t;
typedef __attribute__((ext_vector_type(16))) __bf16 v16bf;
typedef __attribute__((ext_vector_type(8)))  float  fv8;

#define NB 4
#define NC 64
#define NN 256
#define NM 32

struct Frag { v16bf hi; v16bf lo; };

__device__ __forceinline__ void split2(float x, bf16_t& h, bf16_t& l) {
  bf16_t hh = (bf16_t)x;
  h = hh;
  l = (bf16_t)(x - (float)hh);
}

// A fragment (16x32, bf16): lane<16 holds M=lane, K={0..7,16..23};
// lane>=16 holds M=lane-16, K={8..15,24..31} (ISA 7.12.2).
template <typename F>
__device__ __forceinline__ Frag load_frag_a(F f, int lane) {
  Frag r;
  int half = (lane >> 4) & 1, row = lane & 15;
#pragma unroll
  for (int e = 0; e < 16; ++e) {
    int k = half * 8 + (e < 8 ? e : e + 8);
    bf16_t h, l;
    split2(f(row, k), h, l);
    r.hi[e] = h; r.lo[e] = l;
  }
  return r;
}

// B fragment (32x16, bf16): lane<16 -> N=lane, K={0..7,16..23}; lane>=16 -> K={8..15,24..31}.
template <typename F>
__device__ __forceinline__ Frag load_frag_b(F f, int lane) {
  Frag r;
  int half = (lane >> 4) & 1, col = lane & 15;
#pragma unroll
  for (int e = 0; e < 16; ++e) {
    int k = half * 8 + (e < 8 ? e : e + 8);
    bf16_t h, l;
    split2(f(k, col), h, l);
    r.hi[e] = h; r.lo[e] = l;
  }
  return r;
}

// D = A*B + C with split-bf16 correction terms (drops lo*lo).
__device__ __forceinline__ fv8 mma3(const Frag& a, const Frag& b, fv8 c) {
  c = __builtin_amdgcn_wmma_f32_16x16x32_bf16(false, a.hi, false, b.hi, (short)0, c, false, false);
  c = __builtin_amdgcn_wmma_f32_16x16x32_bf16(false, a.hi, false, b.lo, (short)0, c, false, false);
  c = __builtin_amdgcn_wmma_f32_16x16x32_bf16(false, a.lo, false, b.hi, (short)0, c, false, false);
  return c;
}

// D (16x16 f32): VGPR v: lanes<16 -> M=v, N=lane; lanes>=16 -> M=v+8, N=lane-16.
template <typename F>
__device__ __forceinline__ void store_d(fv8 d, F f, int lane) {
  int half = (lane >> 4) & 1, col = lane & 15;
#pragma unroll
  for (int v = 0; v < 8; ++v) f(v + half * 8, col, d[v]);
}

__device__ __forceinline__ float gelu_f(float v) {
  return 0.5f * v * (1.0f + erff(v * 0.7071067811865475f));
}

// ---------------------------------------------------------------------------
// Basis tables: forward DFT cos/sin [32][256]; inverse x-basis invA [2][256][64]
// (real/imag rows vs stacked [Gr;Gi] K); inverse y-basis invY [64][256]
// (stacked [Ur|Ui] K, includes 1/N scaling, x2 Hermitian weights, sign of -sin).
// ---------------------------------------------------------------------------
__global__ void k_bases(float* fwdC, float* fwdS, float* invA, float* invY) {
  int i = blockIdx.x * 256 + threadIdx.x;
  const float w0 = 6.283185307179586f / 256.0f;
  if (i < 32 * 256) {
    int kx = i >> 8, x = i & 255;
    float t = w0 * (float)((kx * x) & 255);
    fwdC[i] = cosf(t);
    fwdS[i] = sinf(t);
  }
  if (i < 2 * 256 * 64) {
    int p = i >> 14, r = i & 16383;
    int x = r >> 6, ks = r & 63, kx = ks & 31;
    float t = w0 * (float)((x * kx) & 255);
    float c = cosf(t) * (1.0f / 256.0f), s = sinf(t) * (1.0f / 256.0f);
    invA[i] = (p == 0) ? ((ks < 32) ? c : -s) : ((ks < 32) ? s : c);
  }
  if (i < 64 * 256) {
    int ks = i >> 8, y = i & 255, ky = ks & 31;
    float a = ((ky == 0) ? 1.0f : 2.0f) / 256.0f;
    float t = w0 * (float)((ky * y) & 255);
    invY[i] = (ks < 32) ? a * cosf(t) : -a * sinf(t);
  }
}

// sw[layer][mode] = sqrt(sum_{i,o} wr^2+wi^2)
__global__ void k_sw(const float* __restrict__ wr, const float* __restrict__ wi,
                     float* __restrict__ sw) {
  int m = blockIdx.x & 1023, layer = blockIdx.x >> 10;
  const float* pr = wr + (size_t)layer * 64 * 64 * 1024;
  const float* pi = wi + (size_t)layer * 64 * 64 * 1024;
  float acc = 0.0f;
  for (int io = threadIdx.x; io < 4096; io += 256) {
    float a = pr[(size_t)io * 1024 + m], b = pi[(size_t)io * 1024 + m];
    acc += a * a + b * b;
  }
  __shared__ float red[256];
  red[threadIdx.x] = acc;
  __syncthreads();
  for (int s = 128; s > 0; s >>= 1) {
    if (threadIdx.x < s) red[threadIdx.x] += red[threadIdx.x + s];
    __syncthreads();
  }
  if (threadIdx.x == 0) sw[blockIdx.x] = sqrtf(red[0]);
}

// wt[m][p][i*64+o] from w[i][o][m]  (per layer)
__global__ void k_trw(const float* __restrict__ wr, const float* __restrict__ wi,
                      float* __restrict__ wt) {
  int idx = blockIdx.x * 256 + threadIdx.x;
  if (idx >= 1024 * 4096) return;
  int m = idx >> 12, io = idx & 4095;
  wt[(size_t)m * 8192 + io]        = wr[(size_t)io * 1024 + m];
  wt[(size_t)m * 8192 + 4096 + io] = wi[(size_t)io * 1024 + m];
}

// Lift: h[b,w,xy] = sum_u Pw[w,u]*x[b,u,xy] + Pb[w]   (K=3 -> VALU)
__global__ void k_lift(const float* __restrict__ x, const float* __restrict__ Pw,
                       const float* __restrict__ Pb, float* __restrict__ h) {
  int i = blockIdx.x * 256 + threadIdx.x;
  if (i >= NB * 65536) return;
  int b = i >> 16, xy = i & 65535;
  float u0 = x[(size_t)(b * 3 + 0) * 65536 + xy];
  float u1 = x[(size_t)(b * 3 + 1) * 65536 + xy];
  float u2 = x[(size_t)(b * 3 + 2) * 65536 + xy];
#pragma unroll 4
  for (int w = 0; w < 64; ++w) {
    float v = Pw[w * 3] * u0 + Pw[w * 3 + 1] * u1 + Pw[w * 3 + 2] * u2 + Pb[w];
    h[((size_t)(b * 64 + w)) * 65536 + xy] = v;
  }
}

// Pointwise 1x1 conv, LDS-staged WMMA GEMM: p[b,o,xy]=sum_i cw[o,i]h[b,i,xy]+cb[o]
__global__ void k_pw(const float* __restrict__ h, const float* __restrict__ cw,
                     const float* __restrict__ cb, float* __restrict__ p) {
  __shared__ float sW[64 * 65];   // cw[o][i], padded
  __shared__ float sH[64 * 33];   // h[i][xy32], padded
  int bid = blockIdx.x;
  int b = bid >> 11, xy0 = (bid & 2047) * 32;
  for (int t = threadIdx.x; t < 4096; t += 256)
    sW[(t >> 6) * 65 + (t & 63)] = cw[t];
  for (int t = threadIdx.x; t < 2048; t += 256) {
    int ii = t >> 5, c = t & 31;
    sH[ii * 33 + c] = h[((size_t)(b * 64 + ii)) * 65536 + xy0 + c];
  }
  __syncthreads();
  int wave = threadIdx.x >> 5, lane = threadIdx.x & 31;
  int o0 = (wave & 3) * 16, n0 = (wave >> 2) * 16;
  fv8 acc = {0, 0, 0, 0, 0, 0, 0, 0};
  for (int k0 = 0; k0 < 64; k0 += 32) {
    Frag A = load_frag_a([&](int m, int k) { return sW[(o0 + m) * 65 + k0 + k]; }, lane);
    Frag B = load_frag_b([&](int k, int n) { return sH[(k0 + k) * 33 + n0 + n]; }, lane);
    acc = mma3(A, B, acc);
  }
  store_d(acc, [&](int m, int n, float v) {
    p[((size_t)(b * 64 + o0 + m)) * 65536 + xy0 + n0 + n] = v + cb[o0 + m];
  }, lane);
}

// Forward DFT stage 1: T[bc][{re,im}][kx][y] = sum_x e^{-i 2pi kx x/N} h[bc,x,y]
__global__ void k_f1(const float* __restrict__ h, const float* __restrict__ fwdC,
                     const float* __restrict__ fwdS, float* __restrict__ T) {
  int bid = blockIdx.x;
  int bc = bid >> 3, yblk = bid & 7;
  int wave = threadIdx.x >> 5, lane = threadIdx.x & 31;
  int kx0 = (wave & 1) * 16;
  int y0 = yblk * 32 + (wave >> 1) * 16;
  const float* hb = h + (size_t)bc * 65536;
  fv8 cr = {0, 0, 0, 0, 0, 0, 0, 0}, ci = {0, 0, 0, 0, 0, 0, 0, 0};
  for (int x0 = 0; x0 < 256; x0 += 32) {
    Frag B  = load_frag_b([&](int k, int n) { return hb[(size_t)(x0 + k) * 256 + y0 + n]; }, lane);
    Frag AC = load_frag_a([&](int m, int k) { return fwdC[(kx0 + m) * 256 + x0 + k]; }, lane);
    Frag AS = load_frag_a([&](int m, int k) { return -fwdS[(kx0 + m) * 256 + x0 + k]; }, lane);
    cr = mma3(AC, B, cr);
    ci = mma3(AS, B, ci);
  }
  float* Tb = T + (size_t)bc * 16384;
  store_d(cr, [&](int m, int n, float v) { Tb[(kx0 + m) * 256 + y0 + n] = v; }, lane);
  store_d(ci, [&](int m, int n, float v) { Tb[8192 + (kx0 + m) * 256 + y0 + n] = v; }, lane);
}

// Forward DFT stage 2: f[bc][{re,im}][kx][ky] = sum_y T[kx,y] e^{-i 2pi ky y/N}
__global__ void k_f2(const float* __restrict__ T, const float* __restrict__ fwdC,
                     const float* __restrict__ fwdS, float* __restrict__ F) {
  int bc = blockIdx.x;
  int wave = threadIdx.x >> 5, lane = threadIdx.x & 31;
  int kx0 = (wave & 1) * 16, ky0 = (wave >> 1) * 16;
  const float* Tr = T + (size_t)bc * 16384;
  const float* Ti = Tr + 8192;
  fv8 fr = {0, 0, 0, 0, 0, 0, 0, 0}, fi = {0, 0, 0, 0, 0, 0, 0, 0};
  for (int y0 = 0; y0 < 256; y0 += 32) {
    Frag Ar  = load_frag_a([&](int m, int k) { return Tr[(kx0 + m) * 256 + y0 + k]; }, lane);
    Frag Ai  = load_frag_a([&](int m, int k) { return Ti[(kx0 + m) * 256 + y0 + k]; }, lane);
    Frag BC  = load_frag_b([&](int k, int n) { return fwdC[(ky0 + n) * 256 + y0 + k]; }, lane);
    Frag BS  = load_frag_b([&](int k, int n) { return fwdS[(ky0 + n) * 256 + y0 + k]; }, lane);
    Frag BSn = load_frag_b([&](int k, int n) { return -fwdS[(ky0 + n) * 256 + y0 + k]; }, lane);
    fr = mma3(Ar, BC, fr); fr = mma3(Ai, BS, fr);   // Tr*cos + Ti*sin
    fi = mma3(Ai, BC, fi); fi = mma3(Ar, BSn, fi);  // Ti*cos - Tr*sin
  }
  float* Fb = F + (size_t)bc * 2048;
  store_d(fr, [&](int m, int n, float v) { Fb[(kx0 + m) * 32 + ky0 + n] = v; }, lane);
  store_d(fi, [&](int m, int n, float v) { Fb[1024 + (kx0 + m) * 32 + ky0 + n] = v; }, lane);
}

// Per-mode complex channel mix (M=4 -> VALU): G[b,o,m] = sw[m]*sum_i f[b,i,m]w[i,o,m]
__global__ void k_mix(const float* __restrict__ F, const float* __restrict__ wt,
                      const float* __restrict__ sw, float* __restrict__ G) {
  int m = blockIdx.x;
  __shared__ float sFr[4][64], sFi[4][64];
  __shared__ float sWr[64 * 64], sWi[64 * 64];
  {
    int t = threadIdx.x, b = t >> 6, i = t & 63;
    sFr[b][i] = F[((size_t)(b * 64 + i)) * 2048 + m];
    sFi[b][i] = F[((size_t)(b * 64 + i)) * 2048 + 1024 + m];
  }
  for (int t = threadIdx.x; t < 4096; t += 256) {
    sWr[t] = wt[(size_t)m * 8192 + t];
    sWi[t] = wt[(size_t)m * 8192 + 4096 + t];
  }
  __syncthreads();
  float s = sw[m];
  int b = threadIdx.x >> 6, o = threadIdx.x & 63;
  float ar = 0.0f, ai = 0.0f;
#pragma unroll 8
  for (int i = 0; i < 64; ++i) {
    float fr = sFr[b][i], fi = sFi[b][i];
    float wr = sWr[i * 64 + o], wi = sWi[i * 64 + o];
    ar += fr * wr - fi * wi;
    ai += fr * wi + fi * wr;
  }
  G[((size_t)(b * 64 + o)) * 2048 + m]        = s * ar;
  G[((size_t)(b * 64 + o)) * 2048 + 1024 + m] = s * ai;
}

// Inverse stage 1: U[bo][x][0..31]=Re, [32..63]=Im of (1/N) sum_kx G e^{+i...}
__global__ void k_i1(const float* __restrict__ G, const float* __restrict__ invA,
                     float* __restrict__ U) {
  int bid = blockIdx.x;
  int bo = bid >> 3, xblk = bid & 7;
  int wave = threadIdx.x >> 5, lane = threadIdx.x & 31;
  int x0 = xblk * 32 + (wave & 1) * 16;
  int ky0 = (wave >> 1) * 16;
  const float* Gb = G + (size_t)bo * 2048;   // [64][32] stacked [Gr;Gi]
  fv8 ur = {0, 0, 0, 0, 0, 0, 0, 0}, ui = {0, 0, 0, 0, 0, 0, 0, 0};
  for (int k0 = 0; k0 < 64; k0 += 32) {
    Frag B  = load_frag_b([&](int k, int n) { return Gb[(k0 + k) * 32 + ky0 + n]; }, lane);
    Frag Ar = load_frag_a([&](int m, int k) { return invA[(x0 + m) * 64 + k0 + k]; }, lane);
    Frag Ai = load_frag_a([&](int m, int k) { return invA[16384 + (x0 + m) * 64 + k0 + k]; }, lane);
    ur = mma3(Ar, B, ur);
    ui = mma3(Ai, B, ui);
  }
  float* Ub = U + (size_t)bo * 16384;
  store_d(ur, [&](int m, int n, float v) { Ub[(x0 + m) * 64 + ky0 + n] = v; }, lane);
  store_d(ui, [&](int m, int n, float v) { Ub[(x0 + m) * 64 + 32 + ky0 + n] = v; }, lane);
}

// Inverse stage 2 + epilogue: h_new = gelu(irfft_y([Ur|Ui]) + p)
__global__ void k_i2(const float* __restrict__ U, const float* __restrict__ invY,
                     float* __restrict__ hio) {
  int bid = blockIdx.x;
  int bo = bid >> 5;
  int xblk = (bid >> 2) & 7, yblk = bid & 3;
  int wave = threadIdx.x >> 5, lane = threadIdx.x & 31;
  int x0 = xblk * 32 + (wave & 1) * 16;
  int y0 = yblk * 64 + (wave >> 1) * 16;
  const float* Ub = U + (size_t)bo * 16384;
  fv8 acc = {0, 0, 0, 0, 0, 0, 0, 0};
  for (int k0 = 0; k0 < 64; k0 += 32) {
    Frag A = load_frag_a([&](int m, int k) { return Ub[(x0 + m) * 64 + k0 + k]; }, lane);
    Frag B = load_frag_b([&](int k, int n) { return invY[(k0 + k) * 256 + y0 + n]; }, lane);
    acc = mma3(A, B, acc);
  }
  float* out = hio + (size_t)bo * 65536;
  store_d(acc, [&](int m, int n, float v) {
    size_t idx = (size_t)(x0 + m) * 256 + y0 + n;
    out[idx] = gelu_f(v + out[idx]);   // out holds pointwise p
  }, lane);
}

// Project: out[b,u,xy] = gelu(sum_w Qw[u,w] h[b,w,xy] + Qb[u])  (M=3 -> VALU)
__global__ void k_proj(const float* __restrict__ h, const float* __restrict__ Qw,
                       const float* __restrict__ Qb, float* __restrict__ out) {
  int i = blockIdx.x * 256 + threadIdx.x;
  if (i >= NB * 65536) return;
  int b = i >> 16, xy = i & 65535;
  float a0 = Qb[0], a1 = Qb[1], a2 = Qb[2];
#pragma unroll 4
  for (int w = 0; w < 64; ++w) {
    float hv = h[((size_t)(b * 64 + w)) * 65536 + xy];
    a0 += Qw[w] * hv;
    a1 += Qw[64 + w] * hv;
    a2 += Qw[128 + w] * hv;
  }
  out[((size_t)(b * 3 + 0)) * 65536 + xy] = gelu_f(a0);
  out[((size_t)(b * 3 + 1)) * 65536 + xy] = gelu_f(a1);
  out[((size_t)(b * 3 + 2)) * 65536 + xy] = gelu_f(a2);
}

// ---------------------------------------------------------------------------
extern "C" void kernel_launch(void* const* d_in, const int* in_sizes, int n_in,
                              void* d_out, int out_size, void* d_ws, size_t ws_size,
                              hipStream_t stream) {
  (void)in_sizes; (void)n_in; (void)out_size;
  const float* x  = (const float*)d_in[0];
  const float* Pw = (const float*)d_in[1];
  const float* Pb = (const float*)d_in[2];
  const float* Qw = (const float*)d_in[3];
  const float* Qb = (const float*)d_in[4];
  const float* wr = (const float*)d_in[5];
  const float* wi = (const float*)d_in[6];
  const float* cw = (const float*)d_in[7];
  const float* cb = (const float*)d_in[8];

  float* ws = (float*)d_ws;
  // Workspace layout (floats); total ~51.5M floats (~196 MB).
  float* fwdC = ws;                       // 8192
  float* fwdS = fwdC + 8192;              // 8192
  float* invA = fwdS + 8192;              // 32768
  float* invY = invA + 32768;             // 16384
  float* swb  = invY + 16384;             // 5120
  float* Fb   = swb + 5120;               // 524288
  float* Gb   = Fb + 524288;              // 524288
  float* Tb   = Gb + 524288;              // 4194304
  float* Ub   = Tb + 4194304;             // 4194304
  float* wt   = Ub + 4194304;             // 8388608
  float* hA   = wt + 8388608;             // 16777216
  float* hB   = hA + 16777216;            // 16777216
  size_t need = (size_t)(hB + 16777216 - ws) * sizeof(float);
  if (ws_size < need) return;             // insufficient scratch

  k_bases<<<128, 256, 0, stream>>>(fwdC, fwdS, invA, invY);
  k_sw<<<5 * 1024, 256, 0, stream>>>(wr, wi, swb);
  k_lift<<<(NB * 65536 + 255) / 256, 256, 0, stream>>>(x, Pw, Pb, hA);

  float* hc = hA;
  float* hn = hB;
  for (int k = 0; k < 5; ++k) {
    const float* wrk = wr + (size_t)k * 64 * 64 * 1024;
    const float* wik = wi + (size_t)k * 64 * 64 * 1024;
    k_trw<<<16384, 256, 0, stream>>>(wrk, wik, wt);
    k_pw<<<8192, 256, 0, stream>>>(hc, cw + (size_t)k * 4096, cb + (size_t)k * 64, hn);
    k_f1<<<2048, 128, 0, stream>>>(hc, fwdC, fwdS, Tb);
    k_f2<<<256, 128, 0, stream>>>(Tb, fwdC, fwdS, Fb);
    k_mix<<<1024, 256, 0, stream>>>(Fb, wt, swb + k * 1024, Gb);
    k_i1<<<2048, 128, 0, stream>>>(Gb, invA, Ub);
    k_i2<<<8192, 256, 0, stream>>>(Ub, invY, hn);
    float* t = hc; hc = hn; hn = t;
  }
  k_proj<<<(NB * 65536 + 255) / 256, 256, 0, stream>>>(hc, Qw, Qb, (float*)d_out);
}
